// Attention_68891275427897
// MI455X (gfx1250) — matrix operand order repaired and checked by reading, not compiled, not run on hardware
//
#include <hip/hip_runtime.h>
#include <hip/hip_bf16.h>

typedef _Float16 h2  __attribute__((ext_vector_type(2)));
typedef _Float16 h8  __attribute__((ext_vector_type(8)));
typedef _Float16 h16 __attribute__((ext_vector_type(16)));
typedef float    f4  __attribute__((ext_vector_type(4)));
typedef float    f8  __attribute__((ext_vector_type(8)));

static constexpr int N = 8192;
static constexpr int D = 64;

// ---------------------------------------------------------------------------
// Cross-half (lane ^ 16) exchange as a pure VALU op: v_permlanex16_b32 with
// identity nibble selectors swaps the two 16-lane halves of a wave32 without
// touching LDS (no ds_bpermute / s_wait_dscnt on the critical path).
// ---------------------------------------------------------------------------
__device__ __forceinline__ int xor16i(int x) {
  return __builtin_amdgcn_permlanex16(x, x, 0x76543210, (int)0xfedcba98, false, false);
}
__device__ __forceinline__ float xor16f(float x) {
  return __builtin_bit_cast(float, xor16i(__builtin_bit_cast(int, x)));
}

// ---------------------------------------------------------------------------
// Prep: fp32 -> fp16. Q is pre-scaled by (1/sqrt(64))*log2(e) so the attention
// kernel uses raw v_exp_f32 (base-2) with no per-score multiply. V is stored
// transposed (VT[v][k]) so the PV WMMA A-operand is a contiguous per-lane load.
// ---------------------------------------------------------------------------
__global__ __launch_bounds__(256) void prep_kernel(
    const float* __restrict__ Q, const float* __restrict__ K,
    const float* __restrict__ V, _Float16* __restrict__ Q16,
    _Float16* __restrict__ K16, _Float16* __restrict__ VT)
{
  int idx = blockIdx.x * blockDim.x + threadIdx.x;
  if (idx >= N * D) return;
  int r = idx >> 6;   // row (q or k index)
  int c = idx & 63;   // d / v-dim
  const float QS = 0.125f * 1.4426950408889634f;
  Q16[idx]      = (_Float16)(Q[idx] * QS);
  K16[idx]      = (_Float16)(K[idx]);
  VT[c * N + r] = (_Float16)(V[idx]);
}

// A-operand loader for V_WMMA_F32_16X16X32_F16 (16Mx32K, 16-bit):
// lanes 0-15 supply K = 0..15 of row M=lane, lanes 16-31 supply
// K = 16..31.  rowp points at (row, K=0) in a row-major f16 matrix.
__device__ __forceinline__ h16 loadA(const _Float16* rowp, int hi) {
  h8 lo = *(const h8*)(rowp + hi * 16);
  h8 hh = *(const h8*)(rowp + hi * 16 + 8);
  return __builtin_shufflevector(lo, hh, 0,1,2,3,4,5,6,7,8,9,10,11,12,13,14,15);
}

// ---------------------------------------------------------------------------
// Flash attention, transposed formulation. One wave per 16-query tile.
//   S^T(16k x 16q) = K_tile(A: 16x32 over d) x Q^T(B: 32x16), acc over d=64
//   O^T(64v x 16q) = V^T(A: 16x32 over keys) x P^T(B: 32x16), 4 v-tiles
// 64 keys per iteration: 8 S WMMAs + 8 PV WMMAs per softmax update.
// ---------------------------------------------------------------------------
__global__ __launch_bounds__(128) void attn_kernel(
    const _Float16* __restrict__ Q16, const _Float16* __restrict__ K16,
    const _Float16* __restrict__ VT, float* __restrict__ out)
{
  const int lane = threadIdx.x & 31;
  const int wave = threadIdx.x >> 5;
  const int col  = lane & 15;   // query column (N-dim of all WMMAs)
  const int hi   = lane >> 4;   // which half of the wave
  const int q0   = (blockIdx.x * 4 + wave) * 16;

  // B-operand Q^T (32x16): lane holds d = hi*16 + 0..15 of query q0+col,
  // 32 contiguous bytes of a Q16 row. Loaded once, lives in regs.
  const _Float16* qrow = Q16 + (size_t)(q0 + col) * D + hi * 16;
  h16 bq0 = *(const h16*)(qrow);        // d  0..31 chunk
  h16 bq1 = *(const h16*)(qrow + 32);   // d 32..63 chunk

  f8 acc[4] = {};
  float m = -__builtin_inff();
  float l = 0.0f;

  for (int jj = 0; jj < N; jj += 64) {
    // ---- S^T for four 16-key tiles (C layout: lane holds keys hi*8+r of col)
    f8 s[4];
    #pragma unroll
    for (int t = 0; t < 4; ++t) {
      const _Float16* kr = K16 + (size_t)(jj + t * 16 + col) * D;
      f8 c = {};
      c = __builtin_amdgcn_wmma_f32_16x16x32_f16(false, loadA(kr,      hi), false, bq0, (short)0, c, false, false);
      c = __builtin_amdgcn_wmma_f32_16x16x32_f16(false, loadA(kr + 32, hi), false, bq1, (short)0, c, false, false);
      s[t] = c;
    }

    // ---- online softmax per query column (column lives in lanes {L, L^16}).
    // Tree-shaped max for ILP: 4 independent per-tile chains, then combine.
    float mt[4];
    #pragma unroll
    for (int t = 0; t < 4; ++t) {
      float a = fmaxf(fmaxf(s[t][0], s[t][1]), fmaxf(s[t][2], s[t][3]));
      float b = fmaxf(fmaxf(s[t][4], s[t][5]), fmaxf(s[t][6], s[t][7]));
      mt[t] = fmaxf(a, b);
    }
    float mloc = fmaxf(fmaxf(mt[0], mt[1]), fmaxf(mt[2], mt[3]));
    mloc = fmaxf(mloc, xor16f(mloc));
    float mnew  = fmaxf(m, mloc);
    float alpha = __builtin_amdgcn_exp2f(m - mnew);   // scores are log2-scaled
    m = mnew;

    float st[4];
    #pragma unroll
    for (int t = 0; t < 4; ++t) {
      #pragma unroll
      for (int r = 0; r < 8; ++r)
        s[t][r] = __builtin_amdgcn_exp2f(s[t][r] - mnew);
      float a = (s[t][0] + s[t][1]) + (s[t][2] + s[t][3]);
      float b = (s[t][4] + s[t][5]) + (s[t][6] + s[t][7]);
      st[t] = a + b;
    }
    l = l * alpha + ((st[0] + st[1]) + (st[2] + st[3]));  // partial; reduced at end
    #pragma unroll
    for (int t = 0; t < 4; ++t)
      #pragma unroll
      for (int r = 0; r < 8; ++r) acc[t][r] *= alpha;

    // ---- two 32-key PV steps; P^T B-operand built with packed-f16 exchanges
    #pragma unroll
    for (int b = 0; b < 2; ++b) {
      // pack this lane's 8 P values of each tile into 4 dwords (2 x f16 each)
      int pk0[4], pk1[4];
      #pragma unroll
      for (int v = 0; v < 4; ++v) {
        pk0[v] = __builtin_bit_cast(int,
                   __builtin_amdgcn_cvt_pkrtz(s[2*b]  [2*v], s[2*b]  [2*v+1]));
        pk1[v] = __builtin_bit_cast(int,
                   __builtin_amdgcn_cvt_pkrtz(s[2*b+1][2*v], s[2*b+1][2*v+1]));
      }
      // B-operand P^T (32x16): lane holds K = hi*16 + 0..15 of its column;
      // the other 8 keys come from the partner lane (L ^ 16), 1 dword / 2 keys.
      union { h16 v; int u[8]; } pb;
      #pragma unroll
      for (int v = 0; v < 4; ++v) {
        int o0 = xor16i(pk0[v]);
        int o1 = xor16i(pk1[v]);
        pb.u[v]     = hi ? o1     : pk0[v];  // K = hi*16 + 2v, 2v+1
        pb.u[4 + v] = hi ? pk1[v] : o0;      // K = hi*16 + 8 + 2v, 2v+1
      }

      // ---- O^T += V^T x P^T over 4 v-tiles; V^T rows contiguous in keys
      #pragma unroll
      for (int t = 0; t < 4; ++t) {
        const _Float16* vr = VT + (size_t)(t * 16 + col) * N + jj + b * 32;
        acc[t] = __builtin_amdgcn_wmma_f32_16x16x32_f16(
                     false, loadA(vr, hi), false, pb.v, (short)0, acc[t], false, false);
      }
    }
  }

  // finish the deferred row-sum reduction and normalize
  l += xor16f(l);
  float inv = 1.0f / l;

  // each lane's 8 values per v-tile are contiguous in memory: vector stores
  float* obase = out + (size_t)(q0 + col) * D + hi * 8;
  #pragma unroll
  for (int t = 0; t < 4; ++t) {
    f4 lo4 = { acc[t][0] * inv, acc[t][1] * inv, acc[t][2] * inv, acc[t][3] * inv };
    f4 hi4 = { acc[t][4] * inv, acc[t][5] * inv, acc[t][6] * inv, acc[t][7] * inv };
    *(f4*)(obase + t * 16)     = lo4;
    *(f4*)(obase + t * 16 + 4) = hi4;
  }
}

extern "C" void kernel_launch(void* const* d_in, const int* in_sizes, int n_in,
                              void* d_out, int out_size, void* d_ws, size_t ws_size,
                              hipStream_t stream) {
  (void)in_sizes; (void)n_in; (void)out_size; (void)ws_size;
  const float* Q = (const float*)d_in[0];
  const float* K = (const float*)d_in[1];
  const float* V = (const float*)d_in[2];
  float* out = (float*)d_out;

  _Float16* Q16 = (_Float16*)d_ws;          // 1 MB
  _Float16* K16 = Q16 + (size_t)N * D;      // 1 MB
  _Float16* VT  = K16 + (size_t)N * D;      // 1 MB (transposed V)

  prep_kernel<<<(N * D) / 256, 256, 0, stream>>>(Q, K, V, Q16, K16, VT);
  attn_kernel<<<N / (16 * 4), 128, 0, stream>>>(Q16, K16, VT, out);
}
